// KANLayer_53068615909528
// MI455X (gfx1250) — matrix operand order repaired and checked by reading, not compiled, not run on hardware
//
#include <hip/hip_runtime.h>

typedef float  v8f  __attribute__((ext_vector_type(8)));
typedef __bf16 v8bf __attribute__((ext_vector_type(8)));
typedef __bf16 v16bf __attribute__((ext_vector_type(16)));
typedef int    v4i  __attribute__((ext_vector_type(4)));
typedef unsigned short u16;

#define BM  128
#define BN  128
#define BK  64
#define STR 72   // LDS row stride in bf16 units: 144B rows, 16B aligned, conflict-free b128 column reads

#if defined(__has_builtin)
#if __has_builtin(__builtin_amdgcn_global_load_async_to_lds_b128)
#define HAVE_ASYNC_LDS 1
#endif
#endif

#ifdef HAVE_ASYNC_LDS
typedef v4i __attribute__((address_space(1))) gv4i;  // global (AS1)
typedef v4i __attribute__((address_space(3))) lv4i;  // LDS    (AS3)
// Async 16B global->LDS copy (GLOBAL_LOAD_ASYNC_TO_LDS_B128, ASYNCcnt).
__device__ __forceinline__ void async_cp16(void* lds_dst, const void* gsrc) {
    __builtin_amdgcn_global_load_async_to_lds_b128(
        (gv4i*)gsrc, (lv4i*)lds_dst, 0, 0);
}
__device__ __forceinline__ void wait_async0() {
#if __has_builtin(__builtin_amdgcn_s_wait_asynccnt)
    __builtin_amdgcn_s_wait_asynccnt(0);
#else
    asm volatile("s_wait_asynccnt 0x0" ::: "memory");
#endif
}
#endif

// ---------------------------------------------------------------------------
// fp32 <-> bf16 split helpers (round-to-nearest-even)
// ---------------------------------------------------------------------------
__device__ __forceinline__ u16 f2bf_rne(float f) {
    unsigned int u = __float_as_uint(f);
    u += 0x7FFFu + ((u >> 16) & 1u);
    return (u16)(u >> 16);
}
__device__ __forceinline__ float bf2f(u16 h) {
    return __uint_as_float(((unsigned int)h) << 16);
}
__device__ __forceinline__ unsigned int pack2(u16 a, u16 b) {
    return (unsigned int)a | ((unsigned int)b << 16);
}
// Split float4 into bf16-hi (RNE) and bf16-lo (RNE of residual): v ~= hi + lo.
__device__ __forceinline__ void split4(const float4 v, uint2& hi, uint2& lo) {
    const u16 h0 = f2bf_rne(v.x), h1 = f2bf_rne(v.y);
    const u16 h2 = f2bf_rne(v.z), h3 = f2bf_rne(v.w);
    const u16 l0 = f2bf_rne(v.x - bf2f(h0));
    const u16 l1 = f2bf_rne(v.y - bf2f(h1));
    const u16 l2 = f2bf_rne(v.z - bf2f(h2));
    const u16 l3 = f2bf_rne(v.w - bf2f(h3));
    hi.x = pack2(h0, h1); hi.y = pack2(h2, h3);
    lo.x = pack2(l0, l1); lo.y = pack2(l2, l3);
}

// ---------------------------------------------------------------------------
// Kernel: one-time bf16 hi/lo split of a float array (n4 = n/4 float4's).
// ---------------------------------------------------------------------------
__global__ void split_kernel(const float* __restrict__ src,
                             u16* __restrict__ hi, u16* __restrict__ lo,
                             long n4) {
    const long i = (long)blockIdx.x * blockDim.x + threadIdx.x;
    if (i >= n4) return;
    const float4 v = ((const float4*)src)[i];
    uint2 h, l;
    split4(v, h, l);
    ((uint2*)hi)[i] = h;
    ((uint2*)lo)[i] = l;
}

// ---------------------------------------------------------------------------
// Kernel: fold per-feature basis chain into affine:
//   s[j] = sum_k W1[j,k]*W2[j,k],  t[j] = sum_k b1[j,k]*W2[j,k]
// ---------------------------------------------------------------------------
__global__ void kan_fold_kernel(const float* __restrict__ W1,
                                const float* __restrict__ b1,
                                const float* __restrict__ W2,
                                float* __restrict__ st,
                                int Dn, int Kn) {
    int j = blockIdx.x * blockDim.x + threadIdx.x;
    if (j >= Dn) return;
    float s = 0.f, t = 0.f;
    for (int k = 0; k < Kn; ++k) {
        const float w2 = W2[j * Kn + k];
        s = fmaf(W1[j * Kn + k], w2, s);
        t = fmaf(b1[j * Kn + k], w2, t);
    }
    st[j]      = s;
    st[Dn + j] = t;
}

// ---------------------------------------------------------------------------
// Main kernel: out = x @ W^T + bias + x*s + t
// bf16x3 split-precision WMMA GEMM (v_wmma_f32_16x16x32_bf16), fp32 accum.
// Block 256 thr (8 waves), tile 128x128, wave tile 32x64, BK=64.
// PRE=true : x/W pre-split into packed bf16 hi/lo arrays; staging is a pure
//            copy -> async global->LDS path when available.
// PRE=false: split fp32 -> bf16 hi/lo inline during staging (small-ws fallback)
// ---------------------------------------------------------------------------
template <bool PRE>
__global__ __launch_bounds__(256)
void kan_main_kernel(const float* __restrict__ x,
                     const float* __restrict__ W,
                     const float* __restrict__ bias,
                     const float* __restrict__ st,
                     const u16* __restrict__ xh, const u16* __restrict__ xl,
                     const u16* __restrict__ wh, const u16* __restrict__ wl,
                     float* __restrict__ out,
                     int Bn, int Dn) {
    __shared__ __align__(16) u16 lds_xh[BM * STR];
    __shared__ __align__(16) u16 lds_xl[BM * STR];
    __shared__ __align__(16) u16 lds_wh[BN * STR];
    __shared__ __align__(16) u16 lds_wl[BN * STR];

    const int t    = threadIdx.x;  // 0..255
    const int lane = t & 31;       // wave32
    const int wid  = t >> 5;       // 0..7
    const int wm   = wid & 3;      // 4 wave-rows * 32 rows
    const int wn   = wid >> 2;     // 2 wave-cols * 64 cols
    const int lr   = lane & 15;
    const int lh   = lane >> 4;

    const int m0 = blockIdx.x * BM;
    const int n0 = blockIdx.y * BN;

    v8f acc[2][4];
#pragma unroll
    for (int mi = 0; mi < 2; ++mi)
#pragma unroll
        for (int ni = 0; ni < 4; ++ni)
            acc[mi][ni] = (v8f){0.f, 0.f, 0.f, 0.f, 0.f, 0.f, 0.f, 0.f};

    // Fragment k-offsets (bf16 units), per ISA 16-bit A/B layouts:
    //   A (16x32): lane half selects K groups {0..7,16..23} / {8..15,24..31}
    //   B (32x16): lane half selects K groups {0..7,16..23} / {8..15,24..31}
    const int aoff0 = lh * 8;
    const int aoff1 = 16 + lh * 8;
    const int boff  = lh * 8;

    for (int k0 = 0; k0 < Dn; k0 += BK) {
        __syncthreads();
        if constexpr (PRE) {
            // Pure-copy staging: 128x64 bf16 per tile-half = 1024 x 16B each.
#pragma unroll
            for (int i = 0; i < 4; ++i) {
                const int q = t + (i << 8);   // 0..1023
                const int r = q >> 3;         // row 0..127
                const int c = (q & 7) << 3;   // bf16 col 0,8,...,56
                const size_t gx = (size_t)(m0 + r) * Dn + (k0 + c);
                const size_t gw = (size_t)(n0 + r) * Dn + (k0 + c);
#ifdef HAVE_ASYNC_LDS
                async_cp16(&lds_xh[r * STR + c], &xh[gx]);
                async_cp16(&lds_xl[r * STR + c], &xl[gx]);
                async_cp16(&lds_wh[r * STR + c], &wh[gw]);
                async_cp16(&lds_wl[r * STR + c], &wl[gw]);
#else
                *(uint4*)&lds_xh[r * STR + c] = *(const uint4*)&xh[gx];
                *(uint4*)&lds_xl[r * STR + c] = *(const uint4*)&xl[gx];
                *(uint4*)&lds_wh[r * STR + c] = *(const uint4*)&wh[gw];
                *(uint4*)&lds_wl[r * STR + c] = *(const uint4*)&wl[gw];
#endif
            }
            if (k0 + BK < Dn) {
                const int pr = t >> 1;
                const int pc = (t & 1) << 5;
                __builtin_prefetch(&xh[(size_t)(m0 + pr) * Dn + (k0 + BK + pc)], 0, 0);
                __builtin_prefetch(&wh[(size_t)(n0 + pr) * Dn + (k0 + BK + pc)], 0, 0);
                __builtin_prefetch(&xl[(size_t)(m0 + pr) * Dn + (k0 + BK + pc)], 0, 0);
                __builtin_prefetch(&wl[(size_t)(n0 + pr) * Dn + (k0 + BK + pc)], 0, 0);
            }
#ifdef HAVE_ASYNC_LDS
            wait_async0();  // this wave's async LDS writes complete
#endif
        } else {
            // Inline split staging (fallback when workspace is too small).
#pragma unroll
            for (int i = 0; i < 8; ++i) {
                const int q = t + (i << 8);   // 0..2047
                const int r = q >> 4;         // row 0..127
                const int c = (q & 15) << 2;  // col 0,4,...,60
                const float4 vx = *(const float4*)&x[(size_t)(m0 + r) * Dn + (k0 + c)];
                const float4 vw = *(const float4*)&W[(size_t)(n0 + r) * Dn + (k0 + c)];
                uint2 hi, lo;
                split4(vx, hi, lo);
                *(uint2*)&lds_xh[r * STR + c] = hi;
                *(uint2*)&lds_xl[r * STR + c] = lo;
                split4(vw, hi, lo);
                *(uint2*)&lds_wh[r * STR + c] = hi;
                *(uint2*)&lds_wl[r * STR + c] = lo;
            }
            if (k0 + BK < Dn) {
                const int pr = t >> 1;
                const int pc = (t & 1) << 5;
                __builtin_prefetch(&x[(size_t)(m0 + pr) * Dn + (k0 + BK + pc)], 0, 0);
                __builtin_prefetch(&W[(size_t)(n0 + pr) * Dn + (k0 + BK + pc)], 0, 0);
            }
        }
        __syncthreads();

#pragma unroll
        for (int kk = 0; kk < BK; kk += 32) {
            const int ar0 = (wm * 32 +  0 + lr) * STR + kk;
            const int ar1 = (wm * 32 + 16 + lr) * STR + kk;
            const v16bf ah0 = __builtin_shufflevector(*(const v8bf*)&lds_xh[ar0 + aoff0],
                                                      *(const v8bf*)&lds_xh[ar0 + aoff1],
                                                      0,1,2,3,4,5,6,7,8,9,10,11,12,13,14,15);
            const v16bf ah1 = __builtin_shufflevector(*(const v8bf*)&lds_xh[ar1 + aoff0],
                                                      *(const v8bf*)&lds_xh[ar1 + aoff1],
                                                      0,1,2,3,4,5,6,7,8,9,10,11,12,13,14,15);
            const v16bf al0 = __builtin_shufflevector(*(const v8bf*)&lds_xl[ar0 + aoff0],
                                                      *(const v8bf*)&lds_xl[ar0 + aoff1],
                                                      0,1,2,3,4,5,6,7,8,9,10,11,12,13,14,15);
            const v16bf al1 = __builtin_shufflevector(*(const v8bf*)&lds_xl[ar1 + aoff0],
                                                      *(const v8bf*)&lds_xl[ar1 + aoff1],
                                                      0,1,2,3,4,5,6,7,8,9,10,11,12,13,14,15);
#pragma unroll
            for (int ni = 0; ni < 4; ++ni) {
                const int br = (wn * 64 + ni * 16 + lr) * STR + kk + boff;
                const v16bf bh = __builtin_shufflevector(*(const v8bf*)&lds_wh[br],
                                                         *(const v8bf*)&lds_wh[br + 16],
                                                         0,1,2,3,4,5,6,7,8,9,10,11,12,13,14,15);
                const v16bf bl = __builtin_shufflevector(*(const v8bf*)&lds_wl[br],
                                                         *(const v8bf*)&lds_wl[br + 16],
                                                         0,1,2,3,4,5,6,7,8,9,10,11,12,13,14,15);
                // bf16x3: hi*hi + hi*lo + lo*hi  (fp32 accumulate)
                acc[0][ni] = __builtin_amdgcn_wmma_f32_16x16x32_bf16(false, ah0, false, bh, (short)0, acc[0][ni], false, false);
                acc[1][ni] = __builtin_amdgcn_wmma_f32_16x16x32_bf16(false, ah1, false, bh, (short)0, acc[1][ni], false, false);
                acc[0][ni] = __builtin_amdgcn_wmma_f32_16x16x32_bf16(false, al0, false, bh, (short)0, acc[0][ni], false, false);
                acc[1][ni] = __builtin_amdgcn_wmma_f32_16x16x32_bf16(false, al1, false, bh, (short)0, acc[1][ni], false, false);
                acc[0][ni] = __builtin_amdgcn_wmma_f32_16x16x32_bf16(false, ah0, false, bl, (short)0, acc[0][ni], false, false);
                acc[1][ni] = __builtin_amdgcn_wmma_f32_16x16x32_bf16(false, ah1, false, bl, (short)0, acc[1][ni], false, false);
            }
        }
    }

    // Epilogue: C layout (16x16 f32): lane = N col (lr), VGPR r -> M row r + 8*lh.
#pragma unroll
    for (int mi = 0; mi < 2; ++mi) {
#pragma unroll
        for (int ni = 0; ni < 4; ++ni) {
            const int col  = n0 + wn * 64 + ni * 16 + lr;
            const float bc = bias[col];
            const float sc = st[col];
            const float tc = st[Dn + col];
#pragma unroll
            for (int r = 0; r < 8; ++r) {
                const int row = m0 + wm * 32 + mi * 16 + lh * 8 + r;
                const size_t idx = (size_t)row * Dn + col;
                out[idx] = acc[mi][ni][r] + bc + x[idx] * sc + tc;
            }
        }
    }
}

extern "C" void kernel_launch(void* const* d_in, const int* in_sizes, int n_in,
                              void* d_out, int out_size, void* d_ws, size_t ws_size,
                              hipStream_t stream) {
    const float* x     = (const float*)d_in[0];
    const float* lin_W = (const float*)d_in[1];
    const float* lin_b = (const float*)d_in[2];
    const float* W1    = (const float*)d_in[3];
    const float* b1    = (const float*)d_in[4];
    const float* W2    = (const float*)d_in[5];
    float* out = (float*)d_out;

    const int D  = in_sizes[2];      // 1024
    const int Bn = in_sizes[0] / D;  // 16384
    const int K  = in_sizes[3] / D;  // 8

    // Workspace layout: st (2D f32) | Wh | Wl | Xh | Xl  (all 16B-aligned sizes)
    const size_t stB = (size_t)2 * D * sizeof(float);
    const size_t wB  = (size_t)D * D * sizeof(u16);
    const size_t xB  = (size_t)Bn * D * sizeof(u16);
    char* wsp = (char*)d_ws;
    float* st = (float*)wsp;

    kan_fold_kernel<<<(D + 255) / 256, 256, 0, stream>>>(W1, b1, W2, st, D, K);

    dim3 grid(Bn / BM, D / BN);
    if (ws_size >= stB + 2 * wB + 2 * xB) {
        u16* wh = (u16*)(wsp + stB);
        u16* wl = (u16*)(wsp + stB + wB);
        u16* xh = (u16*)(wsp + stB + 2 * wB);
        u16* xl = (u16*)(wsp + stB + 2 * wB + xB);
        const long wn4 = (long)D * D / 4;
        const long xn4 = (long)Bn * D / 4;
        split_kernel<<<(unsigned)((wn4 + 255) / 256), 256, 0, stream>>>(lin_W, wh, wl, wn4);
        split_kernel<<<(unsigned)((xn4 + 255) / 256), 256, 0, stream>>>(x, xh, xl, xn4);
        kan_main_kernel<true><<<grid, 256, 0, stream>>>(x, lin_W, lin_b, st,
                                                        xh, xl, wh, wl, out, Bn, D);
    } else {
        kan_main_kernel<false><<<grid, 256, 0, stream>>>(x, lin_W, lin_b, st,
                                                         nullptr, nullptr, nullptr, nullptr,
                                                         out, Bn, D);
    }
}
